// GNN31_46093589020765
// MI455X (gfx1250) — matrix operand order repaired and checked by reading, not compiled, not run on hardware
//
#include <hip/hip_runtime.h>
#include <hip/hip_bf16.h>
#include <math.h>

// ---------------------------------------------------------------------------
// GraphSAGE-pool (3 layers, d=128) for MI455X / gfx1250, wave32 + WMMA bf16.
// Key transform: relu(fc_pool(h))[src] == relu(fc_pool(h[src])) so the pool
// GEMM runs per-node (50k rows), and the per-edge work is only a gather +
// segment-max (atomicMax on int bits, valid since values are >= 0).
// ---------------------------------------------------------------------------

typedef __attribute__((ext_vector_type(16))) __bf16 v16bf;
typedef __attribute__((ext_vector_type(8)))  __bf16 v8bf;
typedef __attribute__((ext_vector_type(4)))  __bf16 v4bf;
typedef __attribute__((ext_vector_type(8)))  float  v8f;

#define N_NODES 50000
#define N_EDGES 800000
#define DIM     128
#define NCLS    10
#define NEG_SLOPE 0.01f

__device__ __forceinline__ __bf16 f2bf(float f) {
    // round-to-nearest-even f32 -> bf16
    unsigned u = __float_as_uint(f);
    u += 0x7FFFu + ((u >> 16) & 1u);
    unsigned short h = (unsigned short)(u >> 16);
    union { unsigned short s; __bf16 b; } cv; cv.s = h;
    return cv.b;
}

// ---- 128x128 weight: fp32 row-major -> bf16 transposed (n-major) ----------
__global__ __launch_bounds__(256)
void wconv_kernel(const float* __restrict__ W, __bf16* __restrict__ Wt) {
    int idx = blockIdx.x * 256 + threadIdx.x;   // 16384 threads
    int k = idx >> 7;
    int n = idx & 127;
    Wt[n * DIM + k] = f2bf(W[k * DIM + n]);
}

// ---- straight fp32 -> bf16 (vectorized x4) --------------------------------
__global__ __launch_bounds__(256)
void bfconv_kernel(const float* __restrict__ in, __bf16* __restrict__ out, int n4) {
    int i = blockIdx.x * 256 + threadIdx.x;
    if (i >= n4) return;
    float4 v = ((const float4*)in)[i];
    v4bf o;
    o[0] = f2bf(v.x); o[1] = f2bf(v.y); o[2] = f2bf(v.z); o[3] = f2bf(v.w);
    ((v4bf*)out)[i] = o;
}

// ---- zero-fill f32 buffer (vectorized x4) ---------------------------------
__global__ __launch_bounds__(256)
void zero_kernel(float4* __restrict__ p, int n4) {
    int i = blockIdx.x * 256 + threadIdx.x;
    if (i < n4) p[i] = make_float4(0.f, 0.f, 0.f, 0.f);
}

// ---------------------------------------------------------------------------
// GEMM: out = act( A1 @ W1t^T  [+ A2 @ W2t^T]  + bias ),  K = N_out = 128.
// A (MxK) bf16 row-major; W*t is the weight stored n-major (Wt[n][k]) so B
// fragments are contiguous loads. One 256-thread block = 128x128 tile,
// wave w -> rows 16w..16w+15, 8 v8f accumulators across the 128 columns.
// All 8 B fragments of a k-step are preloaded into a register array so the
// backend can clause the 16 global_load_b128 together and drain the WMMAs
// with progressively relaxed s_wait_loadcnt (MLP instead of load->wmma
// lockstep).
//
// wave32 WMMA bf16 fragment layouts as loaded here (contiguous k order, same in A and B):
//   A 16x32 : lanes 0-15 row M=lane,     K = k0+[0..15] contiguous
//             lanes16-31 row M=lane-16,  K = k0+[16..31] contiguous
//   B 32x16 : lanes 0-15 col N=lane,     K = k0+[0..15] contiguous
//             lanes16-31 col N=lane-16,  K = k0+[16..31] contiguous
//   C/D 16x16: lane&15 = col, VGPR v = row v (lanes<16) / row 8+v (lanes>=16)
// act: 0 = relu, 1 = leaky_relu(0.01)
// ---------------------------------------------------------------------------
__global__ __launch_bounds__(256)
void gemm128_kernel(const __bf16* __restrict__ A1, const __bf16* __restrict__ W1t,
                    const __bf16* __restrict__ A2, const __bf16* __restrict__ W2t,
                    const float* __restrict__ bias,
                    float* __restrict__ outF, __bf16* __restrict__ outB,
                    int M, int act)
{
    const int tid  = threadIdx.x;
    const int wave = tid >> 5;          // 0..7
    const int lane = tid & 31;
    const int r0   = blockIdx.x * 128 + wave * 16;

    const int mRel = lane & 15;
    const int kSel = (lane >> 4) & 1;
    int aRow = r0 + mRel;
    if (aRow >= M) aRow = M - 1;        // clamp; stores are masked below

    v8f acc[8];
    const v8f vzero = {0.f,0.f,0.f,0.f,0.f,0.f,0.f,0.f};
    #pragma unroll
    for (int i = 0; i < 8; ++i) acc[i] = vzero;

    for (int k0 = 0; k0 < DIM; k0 += 32) {
        // ---- A1 fragment ----
        const __bf16* ap = A1 + (size_t)aRow * DIM + k0 + kSel * 16;
        v8bf alo = *(const v8bf*)ap;           // K = k0+ks   .. +7
        v8bf ahi = *(const v8bf*)(ap + 8);     // K = k0+ks+8 ..+15
        v16bf af;
        #pragma unroll
        for (int i = 0; i < 8; ++i) { af[i] = alo[i]; af[8 + i] = ahi[i]; }

        // ---- preload all 8 B fragments, then issue 8 WMMAs ----
        const __bf16* wbase = W1t + (size_t)mRel * DIM + k0 + kSel * 16;
        v16bf bfr[8];
        #pragma unroll
        for (int nt = 0; nt < 8; ++nt)
            bfr[nt] = *(const v16bf*)(wbase + (size_t)nt * 16 * DIM);
        #pragma unroll
        for (int nt = 0; nt < 8; ++nt)
            acc[nt] = __builtin_amdgcn_wmma_f32_16x16x32_bf16(
                          false, af, false, bfr[nt], (short)0, acc[nt], false, false);

        if (A2) {                               // fused second GEMM (neigh path)
            const __bf16* ap2 = A2 + (size_t)aRow * DIM + k0 + kSel * 16;
            v8bf a2lo = *(const v8bf*)ap2;
            v8bf a2hi = *(const v8bf*)(ap2 + 8);
            v16bf af2;
            #pragma unroll
            for (int i = 0; i < 8; ++i) { af2[i] = a2lo[i]; af2[8 + i] = a2hi[i]; }

            const __bf16* wbase2 = W2t + (size_t)mRel * DIM + k0 + kSel * 16;
            v16bf bfr2[8];
            #pragma unroll
            for (int nt = 0; nt < 8; ++nt)
                bfr2[nt] = *(const v16bf*)(wbase2 + (size_t)nt * 16 * DIM);
            #pragma unroll
            for (int nt = 0; nt < 8; ++nt)
                acc[nt] = __builtin_amdgcn_wmma_f32_16x16x32_bf16(
                              false, af2, false, bfr2[nt], (short)0, acc[nt], false, false);
        }
    }

    // ---- epilogue: bias + activation + store (f32 and/or bf16) ----
    const int cRowBase = (lane >> 4) * 8;       // 0 or 8
    const int cCol     = lane & 15;
    #pragma unroll
    for (int nt = 0; nt < 8; ++nt) {
        const int col = nt * 16 + cCol;
        const float bv = bias ? bias[col] : 0.f;
        #pragma unroll
        for (int v = 0; v < 8; ++v) {
            const int row = r0 + cRowBase + v;
            if (row < M) {
                float val = acc[nt][v] + bv;
                val = (act == 0) ? fmaxf(val, 0.f)
                                 : (val > 0.f ? val : NEG_SLOPE * val);
                const size_t idx = (size_t)row * DIM + col;
                if (outF) outF[idx] = val;
                if (outB) outB[idx] = f2bf(val);
            }
        }
    }
}

// ---------------------------------------------------------------------------
// Edge segment-max: one wave per edge, float4 per lane (128 features).
// P >= 0 (post-relu) and AGG is zero-initialized, so int-bit atomicMax gives
// exactly max(segment_max(P[src]), 0).
// ---------------------------------------------------------------------------
__global__ __launch_bounds__(256)
void edge_max_kernel(const int* __restrict__ src, const int* __restrict__ dst,
                     const float* __restrict__ P, float* __restrict__ AGG, int E)
{
    const int e = blockIdx.x * 8 + (threadIdx.x >> 5);
    if (e >= E) return;
    const int lane = threadIdx.x & 31;
    const int s = src[e];
    const int d = dst[e];
    const float4 v = *(const float4*)(P + (size_t)s * DIM + lane * 4);
    int* out = (int*)(AGG + (size_t)d * DIM + lane * 4);
    atomicMax(out + 0, __float_as_int(v.x));
    atomicMax(out + 1, __float_as_int(v.y));
    atomicMax(out + 2, __float_as_int(v.z));
    atomicMax(out + 3, __float_as_int(v.w));
}

// ---------------------------------------------------------------------------
// Classifier: out = sigmoid(H @ Wout + bout), Wout (128x10) staged in LDS.
// ---------------------------------------------------------------------------
__global__ __launch_bounds__(256)
void classifier_kernel(const float* __restrict__ H, const float* __restrict__ Wout,
                       const float* __restrict__ bout, float* __restrict__ out, int M)
{
    __shared__ float sW[DIM * NCLS];
    __shared__ float sB[NCLS];
    for (int i = threadIdx.x; i < DIM * NCLS; i += 256) sW[i] = Wout[i];
    if (threadIdx.x < NCLS) sB[threadIdx.x] = bout[threadIdx.x];
    __syncthreads();

    const int n = blockIdx.x * 256 + threadIdx.x;
    if (n >= M) return;

    float acc[NCLS];
    #pragma unroll
    for (int c = 0; c < NCLS; ++c) acc[c] = sB[c];

    const float* hrow = H + (size_t)n * DIM;
    for (int k = 0; k < DIM; k += 4) {
        const float4 hv = *(const float4*)(hrow + k);
        #pragma unroll
        for (int c = 0; c < NCLS; ++c) {
            acc[c] += hv.x * sW[(k + 0) * NCLS + c];
            acc[c] += hv.y * sW[(k + 1) * NCLS + c];
            acc[c] += hv.z * sW[(k + 2) * NCLS + c];
            acc[c] += hv.w * sW[(k + 3) * NCLS + c];
        }
    }
    #pragma unroll
    for (int c = 0; c < NCLS; ++c)
        out[(size_t)n * NCLS + c] = 1.f / (1.f + expf(-acc[c]));
}

// ---------------------------------------------------------------------------
extern "C" void kernel_launch(void* const* d_in, const int* in_sizes, int n_in,
                              void* d_out, int out_size, void* d_ws, size_t ws_size,
                              hipStream_t stream)
{
    (void)in_sizes; (void)n_in; (void)out_size; (void)ws_size;

    const float* x   = (const float*)d_in[0];
    const int*   src = (const int*)d_in[1];
    const int*   dst = (const int*)d_in[2];
    const float* Wout = (const float*)d_in[18];
    const float* bout = (const float*)d_in[19];

    // ---- workspace carve-out (256B aligned slabs) ----
    char* ws = (char*)d_ws;
    size_t off = 0;
    auto carve = [&](size_t bytes) -> void* {
        void* p = ws + off;
        off += (bytes + 255) & ~(size_t)255;
        return p;
    };
    __bf16* HbfA  = (__bf16*)carve((size_t)N_NODES * DIM * 2);
    __bf16* HbfB  = (__bf16*)carve((size_t)N_NODES * DIM * 2);
    float*  P     = (float*) carve((size_t)N_NODES * DIM * 4);
    float*  AGG   = (float*) carve((size_t)N_NODES * DIM * 4);
    __bf16* AGGbf = (__bf16*)carve((size_t)N_NODES * DIM * 2);
    float*  Hf    = (float*) carve((size_t)N_NODES * DIM * 4);
    __bf16* Wt[9];
    for (int i = 0; i < 9; ++i) Wt[i] = (__bf16*)carve((size_t)DIM * DIM * 2);

    const int n4_feat = N_NODES * DIM / 4;            // 1,600,000
    const int gConv   = (n4_feat + 255) / 256;        // 6250
    const int gGemm   = (N_NODES + 127) / 128;        // 391
    const int gEdge   = (N_EDGES + 7) / 8;            // 100000
    const int gCls    = (N_NODES + 255) / 256;        // 196

    // ---- prep: x -> bf16, weights -> transposed bf16 ----
    bfconv_kernel<<<gConv, 256, 0, stream>>>(x, HbfA, n4_feat);
    for (int i = 0; i < 3; ++i) {
        const int base = 3 + 5 * i;
        wconv_kernel<<<64, 256, 0, stream>>>((const float*)d_in[base + 0], Wt[3 * i + 0]); // Wpool
        wconv_kernel<<<64, 256, 0, stream>>>((const float*)d_in[base + 2], Wt[3 * i + 1]); // Wself
        wconv_kernel<<<64, 256, 0, stream>>>((const float*)d_in[base + 3], Wt[3 * i + 2]); // Wneigh
    }

    // ---- 3 SAGE-pool layers ----
    __bf16* hcur = HbfA;
    __bf16* hnext = HbfB;
    for (int i = 0; i < 3; ++i) {
        const int base = 3 + 5 * i;
        const float* bpool  = (const float*)d_in[base + 1];
        const float* bneigh = (const float*)d_in[base + 4];

        // P = relu(H @ Wpool + bpool)   (f32 out only)
        gemm128_kernel<<<gGemm, 256, 0, stream>>>(hcur, Wt[3 * i + 0],
                                                  nullptr, nullptr, bpool,
                                                  P, nullptr, N_NODES, /*relu*/0);
        // AGG = max(segment_max(P[src] -> dst), 0)
        zero_kernel<<<gConv, 256, 0, stream>>>((float4*)AGG, n4_feat);
        edge_max_kernel<<<gEdge, 256, 0, stream>>>(src, dst, P, AGG, N_EDGES);
        bfconv_kernel<<<gConv, 256, 0, stream>>>(AGG, AGGbf, n4_feat);

        // H' = leaky_relu(H @ Wself + AGG @ Wneigh + bneigh)  (f32 + bf16 out)
        gemm128_kernel<<<gGemm, 256, 0, stream>>>(hcur, Wt[3 * i + 1],
                                                  AGGbf, Wt[3 * i + 2], bneigh,
                                                  Hf, hnext, N_NODES, /*leaky*/1);
        __bf16* t = hcur; hcur = hnext; hnext = t;
    }

    // ---- classifier head ----
    classifier_kernel<<<gCls, 256, 0, stream>>>(Hf, Wout, bout, (float*)d_out, N_NODES);
}
